// EdgePooling_14637248545217
// MI455X (gfx1250) — matrix operand order repaired and checked by reading, not compiled, not run on hardware
//
#include <hip/hip_runtime.h>
#include <stdint.h>

#define N_NODES  50000
#define IN_CH    128
#define N_EDGES  800000
#define NPAD     (1u << 20)          // edges padded to power of two for bitonic sort
#define ADD_CONST 0.5f

typedef __attribute__((ext_vector_type(16))) __bf16 v16bf;
typedef __attribute__((ext_vector_type(8)))  float  v8f;

union ABf16 { v16bf v; unsigned short h[16]; };

__device__ __forceinline__ unsigned short f32_bf16(float f) {
    union { float f; unsigned u; } x; x.f = f;
    unsigned r = x.u + 0x7FFFu + ((x.u >> 16) & 1u);   // round-to-nearest-even
    return (unsigned short)(r >> 16);
}

// ---------------------------------------------------------------------------
// Kernel 1: dual GEMV p_src = x @ w_src, p_dst = x @ w_dst via WMMA.
// One wave handles a 16-node tile; 4x v_wmma_f32_16x16x32_bf16 over K=128.
// B matrix packs w_src in column 0 and w_dst in column 1 (other cols zero).
// ---------------------------------------------------------------------------
__global__ void proj_wmma(const float* __restrict__ x,
                          const float* __restrict__ wsrc,
                          const float* __restrict__ wdst,
                          float* __restrict__ psrc,
                          float* __restrict__ pdst) {
    const int wave  = (blockIdx.x * blockDim.x + threadIdx.x) >> 5;
    const int lane  = threadIdx.x & 31;
    const int tiles = N_NODES / 16;               // 3125, exact
    if (wave >= tiles) return;                    // wave-uniform branch (EXEC stays all-1)

    const int half  = lane >> 4;                  // K-group half
    const int m     = lane & 15;                  // A row / B,C column
    const int nodeA = wave * 16 + m;
    const float* wcol = (m == 0) ? wsrc : ((m == 1) ? wdst : nullptr);

    v8f acc = {};
    for (int ks = 0; ks < 4; ++ks) {
        const int c0 = ks * 32;
        ABf16 a, b;
        // A: 16-bit A-matrix 16x32 layout — lane holds row m;
        // half=0 -> K {c0..c0+7, c0+16..c0+23}, half=1 -> +8 within each group.
        const float4* xr0 = (const float4*)(x + (size_t)nodeA * IN_CH + c0 + half * 8);
        const float4* xr1 = (const float4*)(x + (size_t)nodeA * IN_CH + c0 + 16 + half * 8);
        float4 a0 = xr0[0], a1 = xr0[1], a2 = xr1[0], a3 = xr1[1];
        const float af[16] = {a0.x,a0.y,a0.z,a0.w, a1.x,a1.y,a1.z,a1.w,
                              a2.x,a2.y,a2.z,a2.w, a3.x,a3.y,a3.z,a3.w};
        #pragma unroll
        for (int i = 0; i < 16; ++i) a.h[i] = f32_bf16(af[i]);

        // B: column m, K {c0+half*8 .. +7, c0+16+half*8 .. +7} (same order as A)
        if (wcol) {
            const float4* wr0 = (const float4*)(wcol + c0 + half * 8);
            const float4* wr1 = (const float4*)(wcol + c0 + 16 + half * 8);
            float4 b0 = wr0[0], b1 = wr0[1], b2 = wr1[0], b3 = wr1[1];
            const float bf[16] = {b0.x,b0.y,b0.z,b0.w, b1.x,b1.y,b1.z,b1.w,
                                  b2.x,b2.y,b2.z,b2.w, b3.x,b3.y,b3.z,b3.w};
            #pragma unroll
            for (int i = 0; i < 16; ++i) b.h[i] = f32_bf16(bf[i]);
        } else {
            #pragma unroll
            for (int i = 0; i < 16; ++i) b.h[i] = 0;
        }

        acc = __builtin_amdgcn_wmma_f32_16x16x32_bf16(
                  /*neg_a=*/false, a.v, /*neg_b=*/false, b.v,
                  /*c_mod=*/(short)0, acc, /*reuse_a=*/false, /*reuse_b=*/false);
    }

    // C/D layout: VGPR r -> M = r + half*8 ; N = lane & 15. We need N==0 (p_src), N==1 (p_dst).
    if (m < 2) {
        float* dst = (m == 0) ? psrc : pdst;
        const int mbase = wave * 16 + half * 8;
        #pragma unroll
        for (int r = 0; r < 8; ++r) dst[mbase + r] = acc[r];
    }
}

// ---------------------------------------------------------------------------
// Workspace init (per node)
// ---------------------------------------------------------------------------
__global__ void init_ws(int* smax, int* haspos, float* denom,
                        int* cluster, float* srep, int* partner, int* repsel) {
    int n = blockIdx.x * blockDim.x + threadIdx.x;
    if (n >= N_NODES) return;
    smax[n] = 0;            // bits of +0.0f; only positive scores inserted
    haspos[n] = 0;
    denom[n] = 0.0f;
    cluster[n] = n;
    srep[n] = 1.0f;
    partner[n] = n;
    repsel[n] = 0;
}

// ---------------------------------------------------------------------------
// Edge score + segment-max of positive scores (int-bit atomicMax is order-
// preserving for positive floats)
// ---------------------------------------------------------------------------
__global__ void edge_score(const float* __restrict__ psrc, const float* __restrict__ pdst,
                           const float* __restrict__ bptr,
                           const int* __restrict__ row, const int* __restrict__ col,
                           float* __restrict__ score, int* smax, int* haspos) {
    int e = blockIdx.x * blockDim.x + threadIdx.x;
    if (e >= N_EDGES) return;
    int r = row[e];
    float s = psrc[r] + pdst[col[e]] + bptr[0];
    score[e] = s;
    if (s > 0.0f) {
        atomicMax(&smax[r], __float_as_int(s));
        atomicOr(&haspos[r], 1);
    }
}

__global__ void edge_denom(const float* __restrict__ score, const int* __restrict__ row,
                           const int* __restrict__ smax, float* denom) {
    int e = blockIdx.x * blockDim.x + threadIdx.x;
    if (e >= N_EDGES) return;
    float s = score[e];
    if (s > 0.0f) {
        int r = row[e];
        atomicAdd(&denom[r], expf(s - __int_as_float(smax[r])));
    }
}

__global__ void edge_norm(const float* __restrict__ score, const int* __restrict__ row,
                          const int* __restrict__ smax, const float* __restrict__ denom,
                          const int* __restrict__ haspos, float* __restrict__ norm) {
    int e = blockIdx.x * blockDim.x + threadIdx.x;
    if (e >= N_EDGES) return;
    float s = score[e];
    int r = row[e];
    float out;
    if (haspos[r]) {
        if (s > 0.0f) {
            float d = denom[r];
            d = (d > 0.0f) ? d : 1.0f;
            out = expf(s - __int_as_float(smax[r])) / d + ADD_CONST;
        } else {
            out = 0.0f;
        }
    } else {
        out = s;
    }
    norm[e] = out;
}

// ---------------------------------------------------------------------------
// Sort: 64-bit keys = (descending-ordered float bits << 32) | edge index.
// Ascending sort of keys == stable descending sort by norm (matches argsort(-norm)).
// ---------------------------------------------------------------------------
__global__ void build_keys(const float* __restrict__ norm, unsigned long long* __restrict__ keys) {
    unsigned i = blockIdx.x * blockDim.x + threadIdx.x;
    if (i >= NPAD) return;
    if (i < N_EDGES) {
        unsigned u = __float_as_uint(norm[i]);
        unsigned asc = (u & 0x80000000u) ? ~u : (u | 0x80000000u);  // total order, ascending
        unsigned dsc = ~asc;                                        // descending
        keys[i] = ((unsigned long long)dsc << 32) | (unsigned long long)i;
    } else {
        keys[i] = 0xFFFFFFFFFFFFFFFFull;                            // pad -> sorts last
    }
}

__global__ void bitonic_step(unsigned long long* __restrict__ keys, unsigned j, unsigned k) {
    unsigned i = blockIdx.x * blockDim.x + threadIdx.x;
    unsigned ixj = i ^ j;
    if (ixj <= i) return;
    unsigned long long a = keys[i];
    unsigned long long b = keys[ixj];
    bool up = ((i & k) == 0);
    if (up ? (a > b) : (a < b)) { keys[i] = b; keys[ixj] = a; }
}

// ---------------------------------------------------------------------------
// Greedy maximal matching in sorted order. Single workgroup; the 50000-node
// matched bitmask (6.25 KB) lives entirely in LDS (320 KB/WGP on CDNA5), so
// the serial scan never touches HBM for state. Chunks of 256 sorted edges are
// loaded cooperatively (coalesced), then resolved in-order by one lane.
// ---------------------------------------------------------------------------
#define NODE_WORDS ((N_NODES + 31) / 32)

__global__ void greedy_match(const unsigned long long* __restrict__ keys,
                             const int* __restrict__ row, const int* __restrict__ col,
                             const float* __restrict__ norm,
                             int* cluster, float* srep, int* partner, int* repsel) {
    __shared__ unsigned matched[NODE_WORDS];
    __shared__ int   su[256];
    __shared__ int   sv[256];
    __shared__ float ss[256];

    const int tid = threadIdx.x;
    for (int w = tid; w < NODE_WORDS; w += blockDim.x) matched[w] = 0u;
    __syncthreads();

    for (unsigned base = 0; base < N_EDGES; base += 256) {   // all real edges sort before pads
        unsigned long long kk = keys[base + tid];
        unsigned idx = (unsigned)(kk & 0xFFFFFFFFull);
        int u = -1, v = -1; float s = 0.0f;
        if (idx < N_EDGES) { u = row[idx]; v = col[idx]; s = norm[idx]; }
        su[tid] = u; sv[tid] = v; ss[tid] = s;
        __syncthreads();
        if (tid == 0) {
            for (int i = 0; i < 256; ++i) {
                int uu = su[i], vv = sv[i];
                if (uu < 0 || uu == vv) continue;
                unsigned wu = (unsigned)uu >> 5, bu = 1u << (uu & 31);
                unsigned wv = (unsigned)vv >> 5, bv = 1u << (vv & 31);
                if (!(matched[wu] & bu) && !(matched[wv] & bv)) {
                    matched[wu] |= bu; matched[wv] |= bv;
                    cluster[vv] = uu;       // v joins cluster of u
                    srep[uu]    = ss[i];    // merge score at representative
                    partner[uu] = vv;
                    repsel[uu]  = 1;
                }
            }
        }
        __syncthreads();
    }
}

// ---------------------------------------------------------------------------
// new_x[n] = sum_x[n]*s_rep[n] + (rep_sel[n] ? x[partner[n]] : 0)
//   rep u        : (x[u]+x[v])*s + x[v]
//   matched v    : 0 (nothing maps to v)
//   unmatched n  : x[n]
// ---------------------------------------------------------------------------
__global__ void new_x_kernel(const float* __restrict__ x, const int* __restrict__ cluster,
                             const float* __restrict__ srep, const int* __restrict__ partner,
                             const int* __restrict__ repsel, float* __restrict__ out) {
    long long t = (long long)blockIdx.x * blockDim.x + threadIdx.x;
    if (t >= (long long)N_NODES * IN_CH) return;
    int n = (int)(t / IN_CH), c = (int)(t % IN_CH);
    float val;
    if (cluster[n] != n) {
        val = 0.0f;
    } else {
        float xv = x[(size_t)n * IN_CH + c];
        if (repsel[n]) {
            float xp = x[(size_t)partner[n] * IN_CH + c];
            val = (xv + xp) * srep[n] + xp;
        } else {
            val = xv * srep[n];       // srep == 1 for unmatched
        }
    }
    out[t] = val;
}

__global__ void node_out_kernel(const int* __restrict__ cluster,
                                float* __restrict__ out_cluster, float* __restrict__ out_isrep) {
    int n = blockIdx.x * blockDim.x + threadIdx.x;
    if (n >= N_NODES) return;
    out_cluster[n] = (float)cluster[n];
    out_isrep[n]   = (cluster[n] == n) ? 1.0f : 0.0f;   // counts>0 <=> not merged-away
}

__global__ void edge_out_kernel(const int* __restrict__ row, const int* __restrict__ col,
                                const int* __restrict__ cluster, const float* __restrict__ ew,
                                float* __restrict__ out_nrow, float* __restrict__ out_ncol,
                                float* __restrict__ out_valid, float* __restrict__ out_ew) {
    int e = blockIdx.x * blockDim.x + threadIdx.x;
    if (e >= N_EDGES) return;
    int nr = cluster[row[e]];
    int nc = cluster[col[e]];
    out_nrow[e]  = (float)nr;
    out_ncol[e]  = (float)nc;
    out_valid[e] = (nr != nc) ? 1.0f : 0.0f;
    out_ew[e]    = ew[e];
}

// ---------------------------------------------------------------------------
extern "C" void kernel_launch(void* const* d_in, const int* in_sizes, int n_in,
                              void* d_out, int out_size, void* d_ws, size_t ws_size,
                              hipStream_t stream) {
    (void)in_sizes; (void)n_in; (void)out_size; (void)ws_size;

    const float* x     = (const float*)d_in[0];
    const float* wsrc  = (const float*)d_in[1];
    const float* wdst  = (const float*)d_in[2];
    const float* bptr  = (const float*)d_in[3];
    const int*   ei    = (const int*)d_in[4];      // [2, E]
    const float* ew    = (const float*)d_in[5];
    const int* row = ei;
    const int* col = ei + N_EDGES;

    // workspace layout (all sizes 16B-multiples)
    char* ws = (char*)d_ws;
    float* psrc   = (float*)ws;                         ws += (size_t)N_NODES * 4;
    float* pdst   = (float*)ws;                         ws += (size_t)N_NODES * 4;
    float* score  = (float*)ws;                         ws += (size_t)N_EDGES * 4;
    int*   smax   = (int*)ws;                           ws += (size_t)N_NODES * 4;
    int*   haspos = (int*)ws;                           ws += (size_t)N_NODES * 4;
    float* denom  = (float*)ws;                         ws += (size_t)N_NODES * 4;
    float* norm   = (float*)ws;                         ws += (size_t)N_EDGES * 4;
    unsigned long long* keys = (unsigned long long*)ws; ws += (size_t)NPAD * 8;
    int*   cluster = (int*)ws;                          ws += (size_t)N_NODES * 4;
    float* srep    = (float*)ws;                        ws += (size_t)N_NODES * 4;
    int*   partner = (int*)ws;                          ws += (size_t)N_NODES * 4;
    int*   repsel  = (int*)ws;                          ws += (size_t)N_NODES * 4;

    // output layout (flat, return order)
    float* out_newx    = (float*)d_out;
    float* out_cluster = out_newx    + (size_t)N_NODES * IN_CH;
    float* out_isrep   = out_cluster + N_NODES;
    float* out_nrow    = out_isrep   + N_NODES;
    float* out_ncol    = out_nrow    + N_EDGES;
    float* out_valid   = out_ncol    + N_EDGES;
    float* out_ew      = out_valid   + N_EDGES;

    const int nodeBlocks = (N_NODES + 255) / 256;
    const int edgeBlocks = (N_EDGES + 255) / 256;

    // 1) projections via WMMA (3125 waves, 8 waves/block)
    proj_wmma<<<(N_NODES / 16 + 7) / 8, 256, 0, stream>>>(x, wsrc, wdst, psrc, pdst);

    // 2) init per-node state
    init_ws<<<nodeBlocks, 256, 0, stream>>>(smax, haspos, denom, cluster, srep, partner, repsel);

    // 3) scores + softmax normalization
    edge_score<<<edgeBlocks, 256, 0, stream>>>(psrc, pdst, bptr, row, col, score, smax, haspos);
    edge_denom<<<edgeBlocks, 256, 0, stream>>>(score, row, smax, denom);
    edge_norm <<<edgeBlocks, 256, 0, stream>>>(score, row, smax, denom, haspos, norm);

    // 4) stable descending sort (bitonic, keys fit in L2)
    build_keys<<<NPAD / 256, 256, 0, stream>>>(norm, keys);
    for (unsigned k = 2; k <= NPAD; k <<= 1)
        for (unsigned j = k >> 1; j > 0; j >>= 1)
            bitonic_step<<<NPAD / 256, 256, 0, stream>>>(keys, j, k);

    // 5) serial greedy matching with LDS-resident bitmask
    greedy_match<<<1, 256, 0, stream>>>(keys, row, col, norm, cluster, srep, partner, repsel);

    // 6) outputs
    new_x_kernel<<<(int)(((long long)N_NODES * IN_CH + 255) / 256), 256, 0, stream>>>(
        x, cluster, srep, partner, repsel, out_newx);
    node_out_kernel<<<nodeBlocks, 256, 0, stream>>>(cluster, out_cluster, out_isrep);
    edge_out_kernel<<<edgeBlocks, 256, 0, stream>>>(row, col, cluster, ew,
                                                    out_nrow, out_ncol, out_valid, out_ew);
}
